// SAGE48_6279242187093
// MI455X (gfx1250) — matrix operand order repaired and checked by reading, not compiled, not run on hardware
//
#include <hip/hip_runtime.h>

// ---------------------------------------------------------------------------
// SAGE48 on gfx1250 (MI455X).
// Aggregation: bf16 gather + fp32 global atomics (bandwidth-bound).
// GEMM: h' = relu([agg*dinv | h] @ [Wl;Wr] + bl) with branch-free
//       v_wmma_f32_16x16x32_bf16 inner loop; A/B pre-packed bf16, fragment
//       loads are contiguous global_load_b128.
// ---------------------------------------------------------------------------

#define NNODES 100000
#define NEDGES 3200000
#define INCH   256
#define NLAYER 48
#define KPADMAX 512   // 2*256

typedef __attribute__((ext_vector_type(16))) __bf16 v16bf;
typedef __attribute__((ext_vector_type(8)))  float  v8f;
typedef __attribute__((ext_vector_type(8)))  unsigned v8u;

__device__ __forceinline__ unsigned short f2bf_u(float f) {
    unsigned u = __builtin_bit_cast(unsigned, f);
    u += 0x7FFFu + ((u >> 16) & 1u);            // round-to-nearest-even
    return (unsigned short)(u >> 16);
}
__device__ __forceinline__ unsigned pk2bf(float a, float b) {
    return (unsigned)f2bf_u(a) | ((unsigned)f2bf_u(b) << 16);
}
__device__ __forceinline__ float bflo(unsigned u) {
    return __builtin_bit_cast(float, u << 16);
}
__device__ __forceinline__ float bfhi(unsigned u) {
    return __builtin_bit_cast(float, u & 0xFFFF0000u);
}

// ---------------- helper kernels ----------------

__global__ void k_zero(float* __restrict__ p, long long n) {
    long long i = (long long)blockIdx.x * blockDim.x + threadIdx.x;
    long long stride = (long long)gridDim.x * blockDim.x;
    for (; i < n; i += stride) p[i] = 0.0f;
}

__global__ void k_deg(const int* __restrict__ dst, float* __restrict__ deg, int E) {
    int e = blockIdx.x * blockDim.x + threadIdx.x;
    if (e < E)
        __hip_atomic_fetch_add(&deg[dst[e]], 1.0f,
                               __ATOMIC_RELAXED, __HIP_MEMORY_SCOPE_AGENT);
}

__global__ void k_deginv(float* __restrict__ deg, int n) {
    int i = blockIdx.x * blockDim.x + threadIdx.x;
    if (i < n) deg[i] = 1.0f / fmaxf(deg[i], 1.0f);
}

// x (f32, N x 256) -> h-half of layer-0 A buffer (bf16, stride 512, offset 256)
__global__ void k_pack_x(const float* __restrict__ x, __bf16* __restrict__ A) {
    long long t = (long long)blockIdx.x * blockDim.x + threadIdx.x;   // N*64
    if (t >= (long long)NNODES * 64) return;
    int n  = (int)(t >> 6);
    int c4 = (int)(t & 63) << 2;
    const float4 v = *(const float4*)(x + (long long)n * INCH + c4);
    uint2 u;
    u.x = pk2bf(v.x, v.y);
    u.y = pk2bf(v.z, v.w);
    *(uint2*)(A + (long long)n * KPADMAX + INCH + c4) = u;
}

// agg (f32) * dinv -> bf16 into A[0..Cin); zero the pad region [2*Cin..Kpad).
__global__ void k_pack_agg(const float* __restrict__ agg,
                           const float* __restrict__ dinv,
                           __bf16* __restrict__ A,
                           int Cin, int Kpad, long long total) {
    long long t = (long long)blockIdx.x * blockDim.x + threadIdx.x;
    if (t >= total) return;
    int cpr = Kpad >> 2;
    int n   = (int)(t / cpr);
    int c4  = (int)(t % cpr) << 2;
    if (c4 >= Cin && c4 < 2 * Cin) return;               // h-half, owned by epilogue
    uint2 u;
    if (c4 < Cin) {
        float di = dinv[n];
        const float4 v = *(const float4*)(agg + (long long)n * Cin + c4);
        u.x = pk2bf(v.x * di, v.y * di);
        u.y = pk2bf(v.z * di, v.w * di);
    } else {
        u.x = 0u; u.y = 0u;                              // K padding
    }
    *(uint2*)(A + (long long)n * Kpad + c4) = u;
}

// Wt[col][k] = bf16 of [Wl;Wr](k, col), zero-padded to Coutp cols / Kpad K.
__global__ void k_pack_w(const float* __restrict__ Wl,
                         const float* __restrict__ Wr,
                         __bf16* __restrict__ Wt,
                         int Cin, int Cout, int Kpad, int Coutp) {
    int t = blockIdx.x * blockDim.x + threadIdx.x;
    int total = Coutp * Kpad;
    if (t >= total) return;
    int col = t / Kpad;
    int k   = t % Kpad;
    float v = 0.0f;
    if (col < Cout) {
        if (k < Cin)            v = Wl[(long long)k * Cout + col];
        else if (k < 2 * Cin)   v = Wr[(long long)(k - Cin) * Cout + col];
    }
    unsigned short s = f2bf_u(v);
    Wt[t] = __builtin_bit_cast(__bf16, s);
}

// gather bf16 h rows (second half of A buffer), fp32-accumulate into agg.
__global__ void k_scatter(const __bf16* __restrict__ h, int stride,
                          const int* __restrict__ src,
                          const int* __restrict__ dst,
                          float* __restrict__ agg,
                          int C, long long total) {
    long long t = (long long)blockIdx.x * blockDim.x + threadIdx.x;
    if (t >= total) return;
    int cg = C >> 2;
    int e  = (int)(t / cg);
    int c4 = (int)(t % cg) << 2;
    int s = src[e], d = dst[e];
    const uint2 u = *(const uint2*)(h + (long long)s * stride + c4);
    float* a = agg + (long long)d * C + c4;
    __hip_atomic_fetch_add(a + 0, bflo(u.x), __ATOMIC_RELAXED, __HIP_MEMORY_SCOPE_AGENT);
    __hip_atomic_fetch_add(a + 1, bfhi(u.x), __ATOMIC_RELAXED, __HIP_MEMORY_SCOPE_AGENT);
    __hip_atomic_fetch_add(a + 2, bflo(u.y), __ATOMIC_RELAXED, __HIP_MEMORY_SCOPE_AGENT);
    __hip_atomic_fetch_add(a + 3, bfhi(u.y), __ATOMIC_RELAXED, __HIP_MEMORY_SCOPE_AGENT);
}

// ---------------- WMMA GEMM ----------------

// A fragment: 8 bf16 at p (K pairs, VGPR0..3) + 8 bf16 at p+16 (VGPR4..7).
__device__ __forceinline__ v16bf ldfragA(const __bf16* p) {
    const uint4 lo = *(const uint4*)(p);
    const uint4 hi = *(const uint4*)(p + 16);
    v8u u = {lo.x, lo.y, lo.z, lo.w, hi.x, hi.y, hi.z, hi.w};
    return __builtin_bit_cast(v16bf, u);
}
// B fragment: 8 bf16 at p + 8 bf16 at p+16 (same K order as A).
__device__ __forceinline__ v16bf ldfragB(const __bf16* p) {
    const uint4 lo = *(const uint4*)(p);
    const uint4 hi = *(const uint4*)(p + 16);
    v8u u = {lo.x, lo.y, lo.z, lo.w, hi.x, hi.y, hi.z, hi.w};
    return __builtin_bit_cast(v16bf, u);
}

// One wave: 16 rows x 64 cols (4 accumulators). Branch-free inner loop.
// Epilogue: bias + ReLU, bf16 store into next layer's A h-half.
__global__ __launch_bounds__(256)
void k_sage_wmma(const __bf16* __restrict__ A, int Kpad,
                 const __bf16* __restrict__ Wt,
                 const float* __restrict__ bl,
                 __bf16* __restrict__ outh, int strideNext,
                 int Cout, int tiles_n, int total_tiles) {
    int wid  = (int)((blockIdx.x * blockDim.x + threadIdx.x) >> 5);
    int lane = threadIdx.x & 31;
    if (wid >= total_tiles) return;                      // wave-uniform exit

    int tm = wid / tiles_n;
    int tn = wid % tiles_n;
    int half = lane >> 4;
    int l15  = lane & 15;

    int a_node = tm * 16 + l15;
    if (a_node >= NNODES) a_node = NNODES - 1;           // clamp, mask at store

    const __bf16* arow = A + (long long)a_node * Kpad + (half << 3);
    int col0 = tn * 64 + l15;                            // cols padded: always valid in Wt
    const __bf16* br0 = Wt + (long long)(col0)      * Kpad + (half << 3);
    const __bf16* br1 = Wt + (long long)(col0 + 16) * Kpad + (half << 3);
    const __bf16* br2 = Wt + (long long)(col0 + 32) * Kpad + (half << 3);
    const __bf16* br3 = Wt + (long long)(col0 + 48) * Kpad + (half << 3);

    v8f c0 = {}, c1 = {}, c2 = {}, c3 = {};

    auto step = [&](int k0) {
        v16bf a  = ldfragA(arow + k0);
        v16bf b0 = ldfragB(br0 + k0);
        v16bf b1 = ldfragB(br1 + k0);
        v16bf b2 = ldfragB(br2 + k0);
        v16bf b3 = ldfragB(br3 + k0);
        c0 = __builtin_amdgcn_wmma_f32_16x16x32_bf16(false, a, false, b0, (short)0, c0, false, false);
        c1 = __builtin_amdgcn_wmma_f32_16x16x32_bf16(false, a, false, b1, (short)0, c1, false, false);
        c2 = __builtin_amdgcn_wmma_f32_16x16x32_bf16(false, a, false, b2, (short)0, c2, false, false);
        c3 = __builtin_amdgcn_wmma_f32_16x16x32_bf16(false, a, false, b3, (short)0, c3, false, false);
    };

    int k0 = 0;
    for (; k0 + 64 <= Kpad; k0 += 64) { step(k0); step(k0 + 32); }
    if (k0 < Kpad) step(k0);

    // epilogue: bias + relu, bf16 stores (masked)
    v8f accs[4] = {c0, c1, c2, c3};
#pragma unroll
    for (int s = 0; s < 4; ++s) {
        int col = tn * 64 + s * 16 + l15;
        float bias = (col < Cout) ? bl[col] : 0.0f;
#pragma unroll
        for (int v = 0; v < 8; ++v) {
            int row = tm * 16 + v + (half << 3);         // D: VGPR v -> M = v + half*8
            if (row < NNODES && col < Cout) {
                float r = accs[s][v] + bias;
                r = (r > 0.0f) ? r : 0.0f;
                unsigned short us = f2bf_u(r);
                outh[(long long)row * strideNext + col] = __builtin_bit_cast(__bf16, us);
            }
        }
    }
}

// final regression: out[n] = sum_j h[n][j]*w[j] + b  (C=4, bf16 h)
__global__ void k_reg(const __bf16* __restrict__ h, int stride,
                      const float* __restrict__ w,
                      const float* __restrict__ b,
                      float* __restrict__ out, int n) {
    int i = blockIdx.x * blockDim.x + threadIdx.x;
    if (i < n) {
        const uint2 u = *(const uint2*)(h + (long long)i * stride);
        out[i] = bflo(u.x) * w[0] + bfhi(u.x) * w[1] +
                 bflo(u.y) * w[2] + bfhi(u.y) * w[3] + b[0];
    }
}

// ---------------- host-side orchestration ----------------

extern "C" void kernel_launch(void* const* d_in, const int* in_sizes, int n_in,
                              void* d_out, int out_size, void* d_ws, size_t ws_size,
                              hipStream_t stream) {
    static const int OUTS[NLAYER] = {
        256,256,256,256,256,256,256,
        128,128,128,128,128,128,128,
        64,64,64,64,64,64,64,
        32,32,32,32,32,32,32,
        16,16,16,16,16,16,16,
        8,8,8,8,8,8,8,
        4,4,4,4,4,4};

    const float* x   = (const float*)d_in[0];
    const int*   ei  = (const int*)d_in[1];
    const int*   src = ei;             // edge_index[0]
    const int*   dst = ei + NEDGES;    // edge_index[1]
    const float* reg_w = (const float*)d_in[2 + 3 * NLAYER];
    const float* reg_b = (const float*)d_in[3 + 3 * NLAYER];

    char* wsb = (char*)d_ws;
    float*  dinv  = (float*)wsb;                               wsb += (size_t)NNODES * 4;
    float*  agg   = (float*)wsb;                               wsb += (size_t)NNODES * INCH * 4;
    __bf16* abufA = (__bf16*)wsb;                              wsb += (size_t)NNODES * KPADMAX * 2;
    __bf16* abufB = (__bf16*)wsb;                              wsb += (size_t)NNODES * KPADMAX * 2;
    __bf16* wt    = (__bf16*)wsb;                              // 256*512 bf16

    // in-degree -> 1/max(deg,1)
    k_zero<<<1024, 256, 0, stream>>>(dinv, (long long)NNODES);
    k_deg<<<(NEDGES + 255) / 256, 256, 0, stream>>>(dst, dinv, NEDGES);
    k_deginv<<<(NNODES + 255) / 256, 256, 0, stream>>>(dinv, NNODES);

    // x -> bf16 h-half of layer-0 A buffer
    {
        long long t = (long long)NNODES * 64;
        k_pack_x<<<(int)((t + 255) / 256), 256, 0, stream>>>(x, abufA);
    }

    int Cin = INCH;
    for (int l = 0; l < NLAYER; ++l) {
        int Cout = OUTS[l];
        const float* Wl = (const float*)d_in[2 + 3 * l];
        const float* bl = (const float*)d_in[3 + 3 * l];
        const float* Wr = (const float*)d_in[4 + 3 * l];

        __bf16* Acur = (l & 1) ? abufB : abufA;
        __bf16* Anxt = (l & 1) ? abufA : abufB;

        int Kpad     = (2 * Cin < 32) ? 32 : 2 * Cin;            // mult of 32
        int CoutNext = Cout;                                     // next layer Cin
        int KpadNext = (2 * CoutNext < 32) ? 32 : 2 * CoutNext;
        int Coutp    = ((Cout + 63) / 64) * 64;                  // wave tile = 64 cols

        // 1) zero agg, scatter-add (gather bf16 h-half of Acur)
        long long aggN = (long long)NNODES * Cin;
        k_zero<<<2048, 256, 0, stream>>>(agg, aggN);
        long long stot = (long long)NEDGES * (Cin >> 2);
        k_scatter<<<(int)((stot + 255) / 256), 256, 0, stream>>>(
            Acur + Cin, Kpad, src, dst, agg, Cin, stot);

        // 2) pack agg*dinv into Acur first half (+ zero pad region)
        long long ptot = (long long)NNODES * (Kpad >> 2);
        k_pack_agg<<<(int)((ptot + 255) / 256), 256, 0, stream>>>(
            agg, dinv, Acur, Cin, Kpad, ptot);

        // 3) pack [Wl;Wr] transposed to bf16
        int wtot = Coutp * Kpad;
        k_pack_w<<<(wtot + 255) / 256, 256, 0, stream>>>(Wl, Wr, wt, Cin, Cout, Kpad, Coutp);

        // 4) fused dual-GEMM + bias + ReLU -> h-half of Anxt
        int tiles_m = (NNODES + 15) / 16;
        int tiles_n = Coutp / 64;
        int tt = tiles_m * tiles_n;
        k_sage_wmma<<<(tt + 7) / 8, 256, 0, stream>>>(
            Acur, Kpad, wt, bl, Anxt + CoutNext, KpadNext, Cout, tiles_n, tt);

        Cin = Cout;
    }

    // final 4 -> 1 regression from last A buffer's h-half (stride 32, offset 4)
    __bf16* Afinal = (NLAYER & 1) ? abufB : abufA;
    k_reg<<<(NNODES + 255) / 256, 256, 0, stream>>>(
        Afinal + 4, 32, reg_w, reg_b, (float*)d_out, NNODES);
}
